// DIENAttentionLayer_67018669687414
// MI455X (gfx1250) — compile-verified, matrix operand order checked by reading, not run on hardware
//
#include <hip/hip_runtime.h>

// Problem constants (from reference): B=2048, T=200, D=128, H=128
#define BB 2048
#define TT 200
#define DD 128
#define HH 128
#define KD 256   // D + H  (GEMM K)
#define NG 256   // 2 * H  (gate GEMM N)

typedef __attribute__((ext_vector_type(16))) __bf16 v16bf;
typedef __attribute__((ext_vector_type(8)))  float  v8f;
typedef __attribute__((ext_vector_type(4)))  unsigned int v4u;
typedef __attribute__((ext_vector_type(8)))  int  v8i;
typedef __attribute__((ext_vector_type(4)))  int  v4i;

union FragU {
    v16bf v;
    uint4 q[2];
};

__device__ __forceinline__ unsigned int f32_to_bf16_raw(float f) {
    unsigned int u = __float_as_uint(f);
    u += 0x7FFFu + ((u >> 16) & 1u);   // round to nearest even
    return u >> 16;
}

__device__ __forceinline__ unsigned short f32_to_bf16(float f) {
    return (unsigned short)f32_to_bf16_raw(f);
}

__device__ __forceinline__ float fast_tanh(float v) {
    float e = __expf(2.0f * v);
    return 1.0f - 2.0f / (e + 1.0f);
}

// ---------------------------------------------------------------------------
// Prep: transpose + downconvert weights to bf16 in [N][K] layout.
// ---------------------------------------------------------------------------
__global__ void augru_prep_weights(const float* __restrict__ Wg,
                                   const float* __restrict__ Wc,
                                   unsigned short* __restrict__ wsg,
                                   unsigned short* __restrict__ wsc) {
    int idx = blockIdx.x * blockDim.x + threadIdx.x;
    if (idx < NG * KD) {
        int n = idx >> 8, k = idx & 255;
        wsg[n * KD + k] = f32_to_bf16(Wg[k * NG + n]);
    }
    if (idx < HH * KD) {
        int n = idx >> 8, k = idx & 255;
        wsc[n * KD + k] = f32_to_bf16(Wc[k * HH + n]);
    }
}

// Fragment load: per-lane K striping per ISA 16-bit layout:
// lanes 0-15 -> K={kk*32+0..7, kk*32+16..23}, lanes 16-31 -> K={kk*32+8..15, kk*32+24..31}.
__device__ __forceinline__ FragU load_frag_row(const unsigned short* rowbase, int kk, int half) {
    const unsigned short* p = rowbase + kk * 32 + half * 8;
    FragU f;
    f.q[0] = *(const uint4*)(p);
    f.q[1] = *(const uint4*)(p + 16);
    return f;
}

// ---------------------------------------------------------------------------
// TDM: async DMA of one 16-row x-tile (16 x 128 fp32, row stride T*D elems)
// from global memory into LDS. 2D descriptor, data_size = 4B.
//   D# group0: count=1 | lds_addr | global_addr | type=2
//   D# group1: data_size=2(4B); tensor_dim0=128; tensor_dim1=16;
//              tile_dim0=128; tile_dim1=16; tensor_dim0_stride=T*D
// Tracked by TENSORcnt; completion enforced with s_wait_tensorcnt.
// (6-arg builtin form: groups 0..3 + extra int32x8 group + cpol)
// ---------------------------------------------------------------------------
__device__ __forceinline__ void tdm_load_x_tile(const float* gsrc, unsigned int lds_byte_addr) {
    unsigned long long ga = (unsigned long long)(const void*)gsrc;
    v4u g0;
    g0.x = 1u;                                            // count=1, user mode
    g0.y = lds_byte_addr;                                 // LDS dest byte address
    g0.z = (unsigned int)(ga & 0xffffffffull);            // global_addr[31:0]
    g0.w = (unsigned int)((ga >> 32) & 0x01ffffffull)     // global_addr[56:32]
         | 0x80000000u;                                   // type = 2 ("image")
    v8i g1;
    g1[0] = (int)(2u << 16);        // workgroup_mask=0 | data_size=2 (4 bytes)
    g1[1] = (int)(128u << 16);      // atomic_barrier=0 | tensor_dim0[15:0]=128
    g1[2] = (int)(16u << 16);       // tensor_dim0[31:16]=0 | tensor_dim1[15:0]=16
    g1[3] = (int)(128u << 16);      // tensor_dim1[31:16]=0 | tile_dim0=128
    g1[4] = 16;                     // tile_dim1=16 | tile_dim2=0
    g1[5] = (int)(TT * DD);         // tensor_dim0_stride[31:0] = 25600
    g1[6] = 0;                      // stride hi | tensor_dim1_stride lo
    g1[7] = 0;
    v4i z4 = {0, 0, 0, 0};          // groups 2/3 unused (2D tensor)
    v8i z8 = {0, 0, 0, 0, 0, 0, 0, 0};
    __builtin_amdgcn_tensor_load_to_lds(g0, g1, z4, z4, z8, 0);
}

// ---------------------------------------------------------------------------
// Persistent AUGRU scan. One block = 16 batch rows, 8 waves (wave32).
// Weight B-fragments live in registers across all T steps; h carried in LDS;
// x tiles DMA'd one step ahead by the Tensor Data Mover (double-buffered).
// ---------------------------------------------------------------------------
__global__ __launch_bounds__(256)
void augru_scan_kernel(const float* __restrict__ x,        // [B,T,D]
                       const float* __restrict__ att,      // [B,T]
                       const int*   __restrict__ seq,      // [B]
                       const unsigned short* __restrict__ wsg, // [256][256] bf16
                       const unsigned short* __restrict__ wsc, // [128][256] bf16
                       const float* __restrict__ bg,       // [256]
                       const float* __restrict__ bc,       // [128]
                       float* __restrict__ out) {          // [B,T,H]
    __shared__ unsigned short Abuf[16 * KD];  // [16][256] bf16: K 0..127 = x, 128..255 = h / r*h
    __shared__ float Hbuf[16 * HH];           // fp32 hidden state
    __shared__ float Ubuf[16 * HH];           // update gate exchange
    __shared__ float Xstage[2][16 * DD];      // TDM double buffer (fp32 x tiles)
    __shared__ float ATT[16];                 // attention scores for this t

    const int tid  = threadIdx.x;
    const int wave = tid >> 5;                // 0..7
    const int lane = tid & 31;
    const int ln   = lane & 15;
    const int half = lane >> 4;
    const int row0 = blockIdx.x * 16;
    const int wv   = __builtin_amdgcn_readfirstlane(wave);  // wave-uniform SGPR

    // ---- loop-invariant per-lane state ----
    float biasg[2];
#pragma unroll
    for (int nt = 0; nt < 2; ++nt) biasg[nt] = bg[wave * 32 + nt * 16 + ln];
    const float biasc = bc[wave * 16 + ln];

    int seqr[8];
#pragma unroll
    for (int j = 0; j < 8; ++j) seqr[j] = seq[row0 + j + half * 8];

    // Weight B-fragments: resident in VGPRs for the whole scan.
    FragU bgf[2][8];
#pragma unroll
    for (int nt = 0; nt < 2; ++nt) {
        const unsigned short* rowb = wsg + (size_t)(wave * 32 + nt * 16 + ln) * KD;
#pragma unroll
        for (int kk = 0; kk < 8; ++kk) bgf[nt][kk] = load_frag_row(rowb, kk, half);
    }
    FragU bcf[8];
    {
        const unsigned short* rowb = wsc + (size_t)(wave * 16 + ln) * KD;
#pragma unroll
        for (int kk = 0; kk < 8; ++kk) bcf[kk] = load_frag_row(rowb, kk, half);
    }

    // ---- h0 = 0 ----
    for (int i = tid; i < 16 * HH; i += 256) {
        Hbuf[i] = 0.0f;
        int m = i >> 7, n = i & 127;
        Abuf[m * KD + DD + n] = 0;  // bf16 +0.0
    }

    // ---- kick off TDM for t = 0 ----
    if (wv == 0) {
        tdm_load_x_tile(x + (size_t)row0 * TT * DD,
                        (unsigned int)(unsigned long long)&Xstage[0][0]);
    }
    __syncthreads();

    const v8f zero8 = {0.f, 0.f, 0.f, 0.f, 0.f, 0.f, 0.f, 0.f};

    // staging assignment: thread handles 8 consecutive floats of one row
    const int sm = tid >> 4;          // row 0..15
    const int sc = (tid & 15) * 8;    // col 0..120

    for (int t = 0; t < TT; ++t) {
        const int cur = t & 1;

        // ---- wait for this step's x tile DMA, then convert fp32 -> bf16 ----
        if (wv == 0) __builtin_amdgcn_s_wait_tensorcnt((short)0);
        __syncthreads();   // publish Xstage[cur] to all waves

        {
            const float* ps = &Xstage[cur][sm * DD + sc];
            float4 f0 = *(const float4*)(ps);
            float4 f1 = *(const float4*)(ps + 4);
            uint4 pk;
            pk.x = f32_to_bf16_raw(f0.x) | (f32_to_bf16_raw(f0.y) << 16);
            pk.y = f32_to_bf16_raw(f0.z) | (f32_to_bf16_raw(f0.w) << 16);
            pk.z = f32_to_bf16_raw(f1.x) | (f32_to_bf16_raw(f1.y) << 16);
            pk.w = f32_to_bf16_raw(f1.z) | (f32_to_bf16_raw(f1.w) << 16);
            *(uint4*)(&Abuf[sm * KD + sc]) = pk;
        }
        if (tid < 16) ATT[tid] = att[(size_t)(row0 + tid) * TT + t];
        __syncthreads();   // Abuf ready; Xstage[cur] consumed

        // ---- prefetch next x tile via TDM while WMMAs run ----
        if (wv == 0 && t + 1 < TT) {
            tdm_load_x_tile(x + ((size_t)row0 * TT + (t + 1)) * DD,
                            (unsigned int)(unsigned long long)&Xstage[(t + 1) & 1][0]);
        }

        // ---- gate GEMM: [16 x 256] @ Wg, wave owns N in [wave*32, wave*32+32) ----
        v8f ga0 = zero8, ga1 = zero8;
        {
            FragU a = load_frag_row(Abuf + ln * KD, 0, half);
#pragma unroll
            for (int kk = 0; kk < 8; ++kk) {
                FragU anext;
                if (kk < 7) anext = load_frag_row(Abuf + ln * KD, kk + 1, half);
                ga0 = __builtin_amdgcn_wmma_f32_16x16x32_bf16(
                    false, a.v, false, bgf[0][kk].v, (short)0, ga0, false, false);
                ga1 = __builtin_amdgcn_wmma_f32_16x16x32_bf16(
                    false, a.v, false, bgf[1][kk].v, (short)0, ga1, false, false);
                a = anext;
            }
        }

        // sigmoid(g + bg)
        float gate[2][8];
#pragma unroll
        for (int j = 0; j < 8; ++j) {
            gate[0][j] = 1.0f / (1.0f + __expf(-(ga0[j] + biasg[0])));
            gate[1][j] = 1.0f / (1.0f + __expf(-(ga1[j] + biasg[1])));
        }
        __syncthreads();  // all waves finished reading A(h) before r*h overwrite

        // waves 0-3 own r (gate cols 0..127): write r*h (bf16) into A[:, 128:256]
        // waves 4-7 own u (gate cols 128..255): park u (fp32) in Ubuf
        if (wave < 4) {
#pragma unroll
            for (int nt = 0; nt < 2; ++nt) {
                int n = wave * 32 + nt * 16 + ln;
#pragma unroll
                for (int j = 0; j < 8; ++j) {
                    int m = j + half * 8;
                    Abuf[m * KD + DD + n] = f32_to_bf16(gate[nt][j] * Hbuf[m * HH + n]);
                }
            }
        } else {
#pragma unroll
            for (int nt = 0; nt < 2; ++nt) {
                int n = (wave - 4) * 32 + nt * 16 + ln;
#pragma unroll
                for (int j = 0; j < 8; ++j) {
                    int m = j + half * 8;
                    Ubuf[m * HH + n] = gate[nt][j];
                }
            }
        }
        __syncthreads();

        // ---- candidate GEMM: [16 x 256] @ Wc, wave owns N in [wave*16, wave*16+16) ----
        v8f ca = zero8;
        {
            FragU a = load_frag_row(Abuf + ln * KD, 0, half);
#pragma unroll
            for (int kk = 0; kk < 8; ++kk) {
                FragU anext;
                if (kk < 7) anext = load_frag_row(Abuf + ln * KD, kk + 1, half);
                ca = __builtin_amdgcn_wmma_f32_16x16x32_bf16(
                    false, a.v, false, bcf[kk].v, (short)0, ca, false, false);
                a = anext;
            }
        }

        // ---- AUGRU update + seq-len masking + output ----
        {
            int n = wave * 16 + ln;
#pragma unroll
            for (int j = 0; j < 8; ++j) {
                int m = j + half * 8;
                float c    = fast_tanh(ca[j] + biasc);
                float u    = ATT[m] * Ubuf[m * HH + n];
                float hold = Hbuf[m * HH + n];
                float hn   = (1.0f - u) * hold + u * c;
                bool valid = t < seqr[j];
                out[((size_t)(row0 + m) * TT + t) * HH + n] = valid ? hn : 0.0f;
                float hk = valid ? hn : hold;
                Hbuf[m * HH + n] = hk;
                Abuf[m * KD + DD + n] = f32_to_bf16(hk);  // h for next step's gate A
            }
        }
        __syncthreads();  // protect A/H/U before next iteration's writes
    }
}

extern "C" void kernel_launch(void* const* d_in, const int* in_sizes, int n_in,
                              void* d_out, int out_size, void* d_ws, size_t ws_size,
                              hipStream_t stream) {
    const float* x   = (const float*)d_in[0];  // inputs [B,T,D]
    const float* att = (const float*)d_in[1];  // att_scores [B,T]
    const int*   seq = (const int*)d_in[2];    // seq_len [B]
    const float* Wg  = (const float*)d_in[3];  // [256,256]
    const float* bg  = (const float*)d_in[4];  // [256]
    const float* Wc  = (const float*)d_in[5];  // [256,128]
    const float* bc  = (const float*)d_in[6];  // [128]
    float* out = (float*)d_out;

    unsigned short* wsg = (unsigned short*)d_ws;      // 256*256 bf16
    unsigned short* wsc = wsg + NG * KD;              // 128*256 bf16

    // 1) weight transpose + bf16 downconvert (one tiny kernel)
    augru_prep_weights<<<(NG * KD + 255) / 256, 256, 0, stream>>>(Wg, Wc, wsg, wsc);

    // 2) persistent WMMA AUGRU scan, weights register-resident, TDM x-prefetch
    augru_scan_kernel<<<BB / 16, 256, 0, stream>>>(x, att, seq, wsg, wsc, bg, bc, out);
}
